// ObjectDetector_31516470018187
// MI455X (gfx1250) — matrix operand order repaired and checked by reading, not compiled, not run on hardware
//
#include <hip/hip_runtime.h>
#include <hip/hip_bf16.h>

typedef __bf16 bf16;
typedef bf16  bf16x8  __attribute__((ext_vector_type(8)));
typedef bf16  bf16x16 __attribute__((ext_vector_type(16)));
typedef float floatx8 __attribute__((ext_vector_type(8)));
typedef unsigned int uint32x4 __attribute__((ext_vector_type(4)));
typedef int  int32x4 __attribute__((ext_vector_type(4)));
typedef int  int32x8 __attribute__((ext_vector_type(8)));

#define B_    64
#define N_    4096
#define DOBS  256
#define D_    128
#define S_    10
#define SP    16      // slots padded to 16 for WMMA
#define HID_  256
#define C_    20
#define E_    64
#define EPS_ATTN 1e-8f

#if defined(__has_builtin)
#if __has_builtin(__builtin_amdgcn_tensor_load_to_lds) && __has_builtin(__builtin_amdgcn_s_wait_tensorcnt)
#define HAVE_TDM 1
#endif
#endif

__device__ __forceinline__ bf16x16 mk16(bf16x8 lo, bf16x8 hi) {
  bf16x16 r;
#pragma unroll
  for (int i = 0; i < 8; i++) { r[i] = lo[i]; r[i + 8] = hi[i]; }
  return r;
}

__device__ __forceinline__ floatx8 wmma_bf16(bf16x16 a, bf16x16 b, floatx8 c) {
  return __builtin_amdgcn_wmma_f32_16x16x32_bf16(false, a, false, b, (short)0, c, false, false);
}

// ---------------- init slots (padded rows 10..15 = 0) ----------------
__global__ void k_init_slots(const float* __restrict__ noise, const float* __restrict__ mu,
                             const float* __restrict__ ls, float* __restrict__ slots) {
  int idx = blockIdx.x * blockDim.x + threadIdx.x;  // B*SP*D
  int d = idx % D_; int s = (idx / D_) % SP; int b = idx / (D_ * SP);
  float v = 0.f;
  if (s < S_) v = mu[d] + __expf(ls[d]) * noise[(b * S_ + s) * D_ + d];
  slots[idx] = v;
}

// ---------------- transpose Wk/Wv to bf16 [out_col][K] ----------------
__global__ void k_prep(const float* __restrict__ Wk, const float* __restrict__ Wv,
                       bf16* __restrict__ Wkt, bf16* __restrict__ Wvt) {
  int i = blockIdx.x * blockDim.x + threadIdx.x;
  if (i < D_ * DOBS) {
    int n = i / DOBS, kk = i % DOBS;
    Wkt[i] = (bf16)Wk[kk * D_ + n];
    Wvt[i] = (bf16)Wv[kk * D_ + n];
  }
}

// ---------------- LN(obs) + K/V projection via WMMA ----------------
// grid: (B*N)/16 blocks x 256 threads (8 waves). Wave w does output col tile w
// for BOTH k and v (8 tiles cover D_=128). obs tile staged to LDS by TDM.
__global__ __launch_bounds__(256) void k_lnproj(
    const float* __restrict__ obs, const float* __restrict__ ni_g, const float* __restrict__ ni_b,
    const bf16* __restrict__ Wkt, const bf16* __restrict__ Wvt,
    bf16* __restrict__ kbf, bf16* __restrict__ vT) {
  __shared__ float xs[16][DOBS];
  __shared__ bf16  xb[16][DOBS];
  __shared__ float part[16][17];
  __shared__ float mres[16], vres[16];
  int t = threadIdx.x;
  long row0 = (long)blockIdx.x * 16;  // global row among B*N (block never crosses batch)

#if defined(HAVE_TDM)
  if (t < 32) {
    // one contiguous 16KB tile: describe as 1-D tensor of 4096 f32 elements
    unsigned lds_base = (unsigned)(unsigned long long)(void*)&xs[0][0];
    unsigned long long ga = (unsigned long long)(const void*)(obs + row0 * (long)DOBS);
    uint32x4 g0;
    g0[0] = 1u;                                    // count=1, user mode
    g0[1] = lds_base;                              // lds_addr
    g0[2] = (unsigned)(ga & 0xFFFFFFFFull);        // global_addr lo
    g0[3] = (unsigned)(ga >> 32) | (2u << 30);     // global_addr hi | type=2 (image)
    int32x8 g1;
    g1[0] = (2 << 16);          // data_size = 4 bytes
    g1[1] = (int)(4096u << 16); // tensor_dim0 = 4096 (low16 @ bits 63:48)
    g1[2] = (1 << 16);          // tensor_dim0 hi16 = 0 | tensor_dim1 = 1
    g1[3] = (int)(4096u << 16); // tensor_dim1 hi16 = 0 | tile_dim0 = 4096
    g1[4] = 1;                  // tile_dim1 = 1, tile_dim2 = 0
    g1[5] = 4096;               // tensor_dim0_stride = 4096 (low32)
    g1[6] = 0;
    g1[7] = 0;
    int32x4 gz = {0, 0, 0, 0};
#if __clang_major__ >= 23
    int32x8 z8 = {0, 0, 0, 0, 0, 0, 0, 0};
    __builtin_amdgcn_tensor_load_to_lds(g0, g1, gz, gz, z8, 0);
#else
    __builtin_amdgcn_tensor_load_to_lds(g0, g1, gz, gz, 0);
#endif
    __builtin_amdgcn_s_wait_tensorcnt(0);
  }
#else
  for (int i = t; i < 16 * DOBS; i += 256) {
    int r = i / DOBS, c = i % DOBS;
    xs[r][c] = obs[(row0 + r) * (long)DOBS + c];
  }
#endif
  __syncthreads();

  // LayerNorm stats: 16 threads per row
  {
    int rr = t >> 4, cc = t & 15;
    float p = 0.f;
#pragma unroll
    for (int c = cc; c < DOBS; c += 16) p += xs[rr][c];
    part[rr][cc] = p;
    __syncthreads();
    if (t < 16) {
      float m = 0.f;
      for (int i = 0; i < 16; i++) m += part[t][i];
      mres[t] = m / (float)DOBS;
    }
    __syncthreads();
    p = 0.f;
#pragma unroll
    for (int c = cc; c < DOBS; c += 16) { float d = xs[rr][c] - mres[rr]; p += d * d; }
    part[rr][cc] = p;
    __syncthreads();
    if (t < 16) {
      float v = 0.f;
      for (int i = 0; i < 16; i++) v += part[t][i];
      vres[t] = rsqrtf(v / (float)DOBS + 1e-5f);
    }
    __syncthreads();
  }
  for (int i = t; i < 16 * DOBS; i += 256) {
    int r = i / DOBS, c = i % DOBS;
    xb[r][c] = (bf16)((xs[r][c] - mres[r]) * vres[r] * ni_g[c] + ni_b[c]);
  }
  __syncthreads();

  int wave = t >> 5, lane = t & 31;
  int r = lane & 15, hi = lane >> 4;
  int n0 = wave * 16;
  floatx8 acck = {}, accv = {};
#pragma unroll
  for (int kk0 = 0; kk0 < DOBS; kk0 += 32) {
    bf16x16 A = *(const bf16x16*)&xb[r][kk0 + hi * 16];
    bf16x16 Bk = *(const bf16x16*)&Wkt[(n0 + r) * DOBS + kk0 + hi * 16];
    bf16x16 Bv = *(const bf16x16*)&Wvt[(n0 + r) * DOBS + kk0 + hi * 16];
    acck = wmma_bf16(A, Bk, acck);
    accv = wmma_bf16(A, Bv, accv);
  }
  long b  = row0 / N_;
  long nb = row0 % N_;
#pragma unroll
  for (int j = 0; j < 8; j++) {
    long rr2 = nb + hi * 8 + j;
    kbf[(b * N_ + rr2) * D_ + n0 + r] = (bf16)acck[j];
  }
  bf16x8 vs;
#pragma unroll
  for (int j = 0; j < 8; j++) vs[j] = (bf16)accv[j];
  *(bf16x8*)&vT[(b * D_ + n0 + r) * (long)N_ + nb + hi * 8] = vs;
}

// ---------------- q = LN(slots)@Wq, pad rows, zero rowsums ----------------
__global__ __launch_bounds__(128) void k_qproj(
    const float* __restrict__ slots, const float* __restrict__ ns_g, const float* __restrict__ ns_b,
    const float* __restrict__ Wq, bf16* __restrict__ qbf, float* __restrict__ rowsum) {
  int b = blockIdx.x, t = threadIdx.x;  // t = d
  __shared__ float sln[S_][D_];
  __shared__ float red[128];
  for (int s = 0; s < S_; s++) {
    float x = slots[(b * SP + s) * D_ + t];
    red[t] = x; __syncthreads();
    for (int o = 64; o > 0; o >>= 1) { if (t < o) red[t] += red[t + o]; __syncthreads(); }
    float m = red[0] / (float)D_; __syncthreads();
    float d0 = x - m; red[t] = d0 * d0; __syncthreads();
    for (int o = 64; o > 0; o >>= 1) { if (t < o) red[t] += red[t + o]; __syncthreads(); }
    float inv = rsqrtf(red[0] / (float)D_ + 1e-5f); __syncthreads();
    sln[s][t] = d0 * inv * ns_g[t] + ns_b[t];
  }
  __syncthreads();
  for (int s = 0; s < SP; s++) {
    float acc = 0.f;
    if (s < S_) for (int k2 = 0; k2 < D_; k2++) acc += sln[s][k2] * Wq[k2 * D_ + t];
    qbf[(b * SP + s) * D_ + t] = (bf16)acc;
  }
  if (t < SP) rowsum[b * SP + t] = 0.f;
}

// ---------------- logits = q@k^T via WMMA, slot-softmax, rowsums ----------------
// grid: B*32 blocks x 256 threads; each wave handles a 16-wide n tile
__global__ __launch_bounds__(256) void k_attn(
    const bf16* __restrict__ qbf, const bf16* __restrict__ kbf,
    bf16* __restrict__ attnb, float* __restrict__ rowsum) {
  int b = blockIdx.x >> 5, cb = blockIdx.x & 31;
  int t = threadIdx.x, wave = t >> 5, lane = t & 31;
  int r = lane & 15, hi = lane >> 4;
  int n0 = cb * 128 + wave * 16;
  const bf16* qb = qbf + b * SP * D_;
  const bf16* kb = kbf + (long)b * N_ * D_;
  floatx8 acc = {};
#pragma unroll
  for (int kk0 = 0; kk0 < D_; kk0 += 32) {
    bf16x16 A = *(const bf16x16*)&qb[r * D_ + kk0 + hi * 16];
    bf16x16 Bm = *(const bf16x16*)&kb[(n0 + r) * (long)D_ + kk0 + hi * 16];
    acc = wmma_bf16(A, Bm, acc);
  }
  __shared__ float tile[8][SP][17];
  const float scale = 0.08838834764831845f;  // 128^-0.5
#pragma unroll
  for (int j = 0; j < 8; j++) tile[wave][hi * 8 + j][r] = acc[j] * scale;
  __syncthreads();
  if (hi == 0) {  // softmax over the 10 real slots, per column r
    float mx = -1e30f;
    for (int s = 0; s < S_; s++) mx = fmaxf(mx, tile[wave][s][r]);
    float sum = 0.f;
    for (int s = 0; s < S_; s++) { float e = __expf(tile[wave][s][r] - mx); tile[wave][s][r] = e; sum += e; }
    float invs = 1.f / sum;
    for (int s = 0; s < S_; s++) tile[wave][s][r] *= invs;
  }
  __syncthreads();
  bf16* ab = attnb + (long)b * SP * N_;
  if (hi == 0) {
    for (int s = 0; s < SP; s++) ab[s * (long)N_ + n0 + r] = (bf16)((s < S_) ? tile[wave][s][r] : 0.f);
  } else if (r < S_) {
    float p = 0.f;
    for (int c = 0; c < 16; c++) p += tile[wave][r][c];
    atomicAdd(&rowsum[b * SP + r], p);
  }
}

// ---------------- partial updates = attn @ v via WMMA; /rowsum applied post-acc ----------------
// grid: B*4 blocks x 256 threads; block cb covers N range [cb*1024, cb*1024+1024)
__global__ __launch_bounds__(256) void k_upd(
    const bf16* __restrict__ attnb, const float* __restrict__ rowsum,
    const bf16* __restrict__ vT, float* __restrict__ updates4) {
  int b = blockIdx.x >> 2, cb = blockIdx.x & 3;
  int t = threadIdx.x, wave = t >> 5, lane = t & 31;
  int r = lane & 15, hi = lane >> 4;
  int d0 = wave * 16;
  const bf16* ab = attnb + (long)b * SP * N_;
  const bf16* vb = vT + (long)b * D_ * N_;
  int nbeg = cb * 1024, nend = nbeg + 1024;
  floatx8 acc = {};
  for (int nn0 = nbeg; nn0 < nend; nn0 += 32) {
    __builtin_prefetch(&vb[(d0 + r) * (long)N_ + nn0 + 256], 0, 0);
    bf16x16 A = *(const bf16x16*)&ab[r * (long)N_ + nn0 + hi * 16];
    bf16x16 Bm = *(const bf16x16*)&vb[(d0 + r) * (long)N_ + nn0 + hi * 16];
    acc = wmma_bf16(A, Bm, acc);
  }
#pragma unroll
  for (int j = 0; j < 8; j++) {
    int s = hi * 8 + j;
    float inv = 1.f / (rowsum[b * SP + s] + EPS_ATTN);
    updates4[(((long)b * 4 + cb) * SP + s) * D_ + d0 + r] = acc[j] * inv;
  }
}

// ---------------- slots += sum(partials); residual MLP ----------------
__global__ __launch_bounds__(128) void k_slotmlp(
    float* __restrict__ slots, const float* __restrict__ updates4,
    const float* __restrict__ nm_g, const float* __restrict__ nm_b,
    const float* __restrict__ W1, const float* __restrict__ b1,
    const float* __restrict__ W2, const float* __restrict__ b2) {
  int b = blockIdx.x, t = threadIdx.x;
  __shared__ float red[128];
  __shared__ float hid[2 * D_];
  for (int s = 0; s < S_; s++) {
    int idx = (b * SP + s) * D_ + t;
    float x = slots[idx];
#pragma unroll
    for (int c = 0; c < 4; c++) x += updates4[(((long)b * 4 + c) * SP + s) * D_ + t];
    red[t] = x; __syncthreads();
    for (int o = 64; o > 0; o >>= 1) { if (t < o) red[t] += red[t + o]; __syncthreads(); }
    float m = red[0] / (float)D_; __syncthreads();
    float dd = x - m; red[t] = dd * dd; __syncthreads();
    for (int o = 64; o > 0; o >>= 1) { if (t < o) red[t] += red[t + o]; __syncthreads(); }
    float inv = rsqrtf(red[0] / (float)D_ + 1e-5f); __syncthreads();
    float h = dd * inv * nm_g[t] + nm_b[t];
    red[t] = h; __syncthreads();               // red now holds LN(slots)
#pragma unroll
    for (int u = 0; u < 2; u++) {
      int j = t * 2 + u;
      float a = b1[j];
      for (int k2 = 0; k2 < D_; k2++) a += red[k2] * W1[k2 * 2 * D_ + j];
      hid[j] = fmaxf(a, 0.f);
    }
    __syncthreads();
    float o2 = b2[t];
    for (int j = 0; j < 2 * D_; j++) o2 += hid[j] * W2[j * D_ + t];
    slots[idx] = x + o2;
    __syncthreads();
  }
}

// ---------------- output heads ----------------
__global__ __launch_bounds__(128) void k_heads(
    const float* __restrict__ slots,
    const float* __restrict__ pdW1, const float* __restrict__ pdb1,
    const float* __restrict__ pdW2, const float* __restrict__ pdb2,
    const float* __restrict__ clsW, const float* __restrict__ clsb,
    const float* __restrict__ cemb, const float* __restrict__ temp,
    const float* __restrict__ bbW1, const float* __restrict__ bbb1,
    const float* __restrict__ bbW2, const float* __restrict__ bbb2,
    float* __restrict__ out_obj, float* __restrict__ out_types, float* __restrict__ out_bb) {
  int b = blockIdx.x, t = threadIdx.x;
  __shared__ float sl[D_];
  __shared__ float hid[HID_];
  __shared__ float feat[E_];
  __shared__ float cn[C_][E_];
  __shared__ float fnrm;
  for (int i = t; i < C_ * E_; i += 128) cn[i / E_][i % E_] = cemb[i];
  __syncthreads();
  if (t < C_) {
    float nrm = 0.f;
    for (int e = 0; e < E_; e++) nrm += cn[t][e] * cn[t][e];
    nrm = fmaxf(sqrtf(nrm), 1e-12f);
    for (int e = 0; e < E_; e++) cn[t][e] /= nrm;
  }
  __syncthreads();
  float tmp = temp[0];
  for (int s = 0; s < S_; s++) {
    sl[t] = slots[(b * SP + s) * D_ + t];
    __syncthreads();
    // objects head
#pragma unroll
    for (int u = 0; u < 2; u++) {
      int j = t * 2 + u;
      float a = pdb1[j];
      for (int k2 = 0; k2 < D_; k2++) a += sl[k2] * pdW1[k2 * HID_ + j];
      hid[j] = fmaxf(a, 0.f);
    }
    __syncthreads();
    float o2 = pdb2[t];
    for (int j = 0; j < HID_; j++) o2 += hid[j] * pdW2[j * D_ + t];
    out_obj[(b * S_ + s) * D_ + t] = o2;
    __syncthreads();
    // class head
    if (t < E_) {
      float a = clsb[t];
      for (int k2 = 0; k2 < D_; k2++) a += sl[k2] * clsW[k2 * E_ + t];
      feat[t] = a;
    }
    __syncthreads();
    if (t == 0) {
      float nrm = 0.f;
      for (int e = 0; e < E_; e++) nrm += feat[e] * feat[e];
      fnrm = fmaxf(sqrtf(nrm), 1e-12f);
    }
    __syncthreads();
    if (t < C_) {
      float a = 0.f;
      for (int e = 0; e < E_; e++) a += (feat[e] / fnrm) * cn[t][e];
      out_types[(b * S_ + s) * C_ + t] = a / tmp;
    }
    __syncthreads();
    // bbox head
#pragma unroll
    for (int u = 0; u < 2; u++) {
      int j = t * 2 + u;
      float a = bbb1[j];
      for (int k2 = 0; k2 < D_; k2++) a += sl[k2] * bbW1[k2 * HID_ + j];
      hid[j] = fmaxf(a, 0.f);
    }
    __syncthreads();
    if (t < 4) {
      float a = bbb2[t];
      for (int j = 0; j < HID_; j++) a += hid[j] * bbW2[j * 4 + t];
      out_bb[(b * S_ + s) * 4 + t] = 1.f / (1.f + __expf(-a));
    }
    __syncthreads();
  }
}

static inline size_t align256(size_t x) { return (x + 255) & ~(size_t)255; }

extern "C" void kernel_launch(void* const* d_in, const int* in_sizes, int n_in,
                              void* d_out, int out_size, void* d_ws, size_t ws_size,
                              hipStream_t stream) {
  (void)in_sizes; (void)n_in; (void)out_size; (void)ws_size;
  const float* obs   = (const float*)d_in[0];
  const float* noise = (const float*)d_in[1];
  const float* mu    = (const float*)d_in[2];
  const float* lsg   = (const float*)d_in[3];
  const float* ni_g  = (const float*)d_in[4];
  const float* ni_b  = (const float*)d_in[5];
  const float* ns_g  = (const float*)d_in[6];
  const float* ns_b  = (const float*)d_in[7];
  const float* nm_g  = (const float*)d_in[8];
  const float* nm_b  = (const float*)d_in[9];
  const float* Wq    = (const float*)d_in[10];
  const float* Wk    = (const float*)d_in[11];
  const float* Wv    = (const float*)d_in[12];
  const float* mW1   = (const float*)d_in[13];
  const float* mb1   = (const float*)d_in[14];
  const float* mW2   = (const float*)d_in[15];
  const float* mb2   = (const float*)d_in[16];
  const float* pdW1  = (const float*)d_in[17];
  const float* pdb1  = (const float*)d_in[18];
  const float* pdW2  = (const float*)d_in[19];
  const float* pdb2  = (const float*)d_in[20];
  const float* clsW  = (const float*)d_in[21];
  const float* clsb  = (const float*)d_in[22];
  const float* cemb  = (const float*)d_in[23];
  const float* temp  = (const float*)d_in[24];
  const float* bbW1  = (const float*)d_in[25];
  const float* bbb1  = (const float*)d_in[26];
  const float* bbW2  = (const float*)d_in[27];
  const float* bbb2  = (const float*)d_in[28];

  // workspace carve-up
  char* ws = (char*)d_ws;
  size_t off = 0;
  bf16*  kbf      = (bf16*) (ws + off); off = align256(off + (size_t)B_ * N_ * D_ * sizeof(bf16));
  bf16*  vT       = (bf16*) (ws + off); off = align256(off + (size_t)B_ * D_ * N_ * sizeof(bf16));
  bf16*  Wkt      = (bf16*) (ws + off); off = align256(off + (size_t)D_ * DOBS * sizeof(bf16));
  bf16*  Wvt      = (bf16*) (ws + off); off = align256(off + (size_t)D_ * DOBS * sizeof(bf16));
  float* slots    = (float*)(ws + off); off = align256(off + (size_t)B_ * SP * D_ * sizeof(float));
  bf16*  qbf      = (bf16*) (ws + off); off = align256(off + (size_t)B_ * SP * D_ * sizeof(bf16));
  bf16*  attnb    = (bf16*) (ws + off); off = align256(off + (size_t)B_ * SP * N_ * sizeof(bf16));
  float* rowsum   = (float*)(ws + off); off = align256(off + (size_t)B_ * SP * sizeof(float));
  float* updates4 = (float*)(ws + off); off = align256(off + (size_t)B_ * 4 * SP * D_ * sizeof(float));

  float* out_obj   = (float*)d_out;
  float* out_types = out_obj + (size_t)B_ * S_ * D_;
  float* out_bb    = out_types + (size_t)B_ * S_ * C_;

  k_prep<<<(D_ * DOBS + 255) / 256, 256, 0, stream>>>(Wk, Wv, Wkt, Wvt);
  k_init_slots<<<(B_ * SP * D_) / 256, 256, 0, stream>>>(noise, mu, lsg, slots);
  k_lnproj<<<(B_ * N_) / 16, 256, 0, stream>>>(obs, ni_g, ni_b, Wkt, Wvt, kbf, vT);

  for (int it = 0; it < 3; it++) {
    k_qproj<<<B_, 128, 0, stream>>>(slots, ns_g, ns_b, Wq, qbf, rowsum);
    k_attn<<<B_ * 32, 256, 0, stream>>>(qbf, kbf, attnb, rowsum);
    k_upd<<<B_ * 4, 256, 0, stream>>>(attnb, rowsum, vT, updates4);
    k_slotmlp<<<B_, 128, 0, stream>>>(slots, updates4, nm_g, nm_b, mW1, mb1, mW2, mb2);
  }

  k_heads<<<B_, 128, 0, stream>>>(slots, pdW1, pdb1, pdW2, pdb2, clsW, clsb, cemb, temp,
                                  bbW1, bbb1, bbW2, bbb2, out_obj, out_types, out_bb);
}
